// HeteroGCN_6743098655603
// MI455X (gfx1250) — matrix operand order repaired and checked by reading, not compiled, not run on hardware
//
#include <hip/hip_runtime.h>
#include <hip/hip_bf16.h>

// ---------------------------------------------------------------------------
// HeteroGCN (2-layer) for MI455X / gfx1250, wave32 + WMMA bf16->f32.
// Layer 0 collapses analytically (uniform seed embeddings -> degree scaling).
// Layer 1: persistent-wave WMMA edge-message GEMMs (weights register-resident)
// with f32 atomic scatter + WMMA node update over the concat features.
// ---------------------------------------------------------------------------

typedef __bf16 bf16x16 __attribute__((ext_vector_type(16)));
typedef float  f32x8   __attribute__((ext_vector_type(8)));

#define NA 100000
#define NB 100000
#define EAA 400000
#define EAB 800000
#define EBA 800000
#define DD 64

union AFrag { uint4 u[2]; bf16x16 v; };

static __device__ __forceinline__ unsigned short f2bf(float f) {
    unsigned int u = __float_as_uint(f);
    // round-to-nearest-even bf16
    unsigned int r = (u + 0x7FFFu + ((u >> 16) & 1u)) >> 16;
    return (unsigned short)r;
}

// --------------------------- utility kernels -------------------------------

__global__ void zero_u32_kernel(unsigned int* p, long long n) {
    long long i = (long long)blockIdx.x * blockDim.x + threadIdx.x;
    long long stride = (long long)gridDim.x * blockDim.x;
    for (; i < n; i += stride) p[i] = 0u;
}

__global__ void degree_kernel(const int* __restrict__ dst, int* __restrict__ deg, int E) {
    int i = blockIdx.x * blockDim.x + threadIdx.x;
    if (i < E) atomicAdd(&deg[dst[i]], 1);
}

// Pack a [KT*32, 64] f32 weight matrix into WMMA bf16 B-fragment layout:
// out[kt][nt][lane][j] = W[(kt*32 + (j>>3)*16 + (lane>>4)*8 + (j&7))*64 + nt*16 + (lane&15)]
__global__ void pack_w_kernel(const float* __restrict__ W, unsigned short* __restrict__ out, int KT) {
    int idx = blockIdx.x * blockDim.x + threadIdx.x;
    int total = KT * 2048;
    if (idx >= total) return;
    int j    = idx & 15;
    int lane = (idx >> 4) & 31;
    int nt   = (idx >> 9) & 3;
    int kt   = idx >> 11;
    int k = kt * 32 + (j >> 3) * 16 + (lane >> 4) * 8 + (j & 7);
    int n = nt * 16 + (lane & 15);
    out[idx] = f2bf(W[k * 64 + n]);
}

// --------------------------- layer-0 (analytic) ----------------------------
// vecs layout (floats): [0]=maa [64]=mab [128]=mba [192]=cA [256]=uA [320]=vA
//                       [384]=cB [448]=uB
__global__ void layer0_precompute_kernel(
    const float* __restrict__ embA, const float* __restrict__ embB,
    const float* __restrict__ Waa0, const float* __restrict__ baa0,
    const float* __restrict__ Wab0, const float* __restrict__ bab0,
    const float* __restrict__ Wba0, const float* __restrict__ bba0,
    const float* __restrict__ WnA0, const float* __restrict__ bnA0,
    const float* __restrict__ WnB0, const float* __restrict__ bnB0,
    float* __restrict__ vecs)
{
    int j = threadIdx.x;  // 0..63
    float saa = baa0[j], sab = bab0[j], sba = bba0[j];
    for (int k = 0; k < DD; ++k) {
        saa += embA[k] * Waa0[k * 64 + j];
        sab += embA[k] * Wab0[k * 64 + j];
        sba += embB[k] * Wba0[k * 64 + j];
    }
    vecs[0 * 64 + j] = fmaxf(saa, 0.f);   // maa
    vecs[1 * 64 + j] = fmaxf(sab, 0.f);   // mab
    vecs[2 * 64 + j] = fmaxf(sba, 0.f);   // mba
    __syncthreads();
    float cA = bnA0[j], uA = 0.f, vA = 0.f, cB = bnB0[j], uB = 0.f;
    for (int k = 0; k < DD; ++k) {
        cA += embA[k]          * WnA0[(k)       * 64 + j];
        uA += vecs[0 * 64 + k] * WnA0[(64 + k)  * 64 + j];
        vA += vecs[2 * 64 + k] * WnA0[(128 + k) * 64 + j];
        cB += embB[k]          * WnB0[(k)       * 64 + j];
        uB += vecs[1 * 64 + k] * WnB0[(64 + k)  * 64 + j];
    }
    vecs[3 * 64 + j] = cA;
    vecs[4 * 64 + j] = uA;
    vecs[5 * 64 + j] = vA;
    vecs[6 * 64 + j] = cB;
    vecs[7 * 64 + j] = uB;
}

__global__ void layer0_nodeA_kernel(const float* __restrict__ vecs,
                                    const int* __restrict__ degaa, const int* __restrict__ degba,
                                    unsigned short* __restrict__ h1, int N)
{
    int i = blockIdx.x * blockDim.x + threadIdx.x;
    if (i >= N * 64) return;
    int n = i >> 6, j = i & 63;
    float v = vecs[192 + j] + (float)degaa[n] * vecs[256 + j] + (float)degba[n] * vecs[320 + j];
    h1[i] = f2bf(fmaxf(v, 0.f));
}

__global__ void layer0_nodeB_kernel(const float* __restrict__ vecs,
                                    const int* __restrict__ degab,
                                    unsigned short* __restrict__ h1, int N)
{
    int i = blockIdx.x * blockDim.x + threadIdx.x;
    if (i >= N * 64) return;
    int n = i >> 6, j = i & 63;
    float v = vecs[384 + j] + (float)degab[n] * vecs[448 + j];
    h1[i] = f2bf(fmaxf(v, 0.f));
}

// --------------------------- layer-1 edge messages -------------------------
// Persistent waves. B fragments (2 kt x 4 nt) live in registers for the whole
// kernel; each wave grid-strides over 16-edge tiles:
//   gather A (bf16) -> 8 back-to-back WMMAs -> bias+relu -> f32 atomic scatter.
// E is a multiple of 16 (asserted by launch).
__global__ __launch_bounds__(256) void edge_msg_kernel(
    const unsigned short* __restrict__ hsrc,   // bf16 [Nsrc*64]
    const int* __restrict__ src, const int* __restrict__ dst,
    const unsigned short* __restrict__ Wpk,    // bf16 [2][4][32][16]
    const float* __restrict__ bias,            // [64]
    float* __restrict__ racc,                  // f32 [Ndst*64]
    int tiles)                                 // E/16
{
    __shared__ unsigned short wlds[4096];      // 2 kt * 4 nt * 32 lanes * 16
    for (int i = threadIdx.x; i < 4096; i += 256)
        wlds[i] = Wpk[i];
    __syncthreads();

    int lane = threadIdx.x & 31;
    int row  = lane & 15;
    int hi   = lane >> 4;

    // register-resident B fragments
    AFrag b[8];
    #pragma unroll
    for (int f = 0; f < 8; ++f) {
        const unsigned short* wp = &wlds[(f * 32 + lane) << 4];
        b[f].u[0] = *(const uint4*)(wp);
        b[f].u[1] = *(const uint4*)(wp + 8);
    }
    float bv[4];
    #pragma unroll
    for (int nt = 0; nt < 4; ++nt) bv[nt] = bias[nt * 16 + row];

    int wave   = blockIdx.x * (blockDim.x >> 5) + (threadIdx.x >> 5);
    int nwaves = gridDim.x * (blockDim.x >> 5);
    int k0 = hi * 8;

    for (int tile = wave; tile < tiles; tile += nwaves) {   // wave-uniform loop
        int ebase = tile * 16;
        const unsigned short* hp = hsrc + (long long)src[ebase + row] * 64;
        AFrag a0, a1;
        a0.u[0] = *(const uint4*)(hp + k0);            // K k0..k0+7
        a0.u[1] = *(const uint4*)(hp + k0 + 16);       // K k0+16..k0+23
        a1.u[0] = *(const uint4*)(hp + k0 + 32);
        a1.u[1] = *(const uint4*)(hp + k0 + 48);
        int myDst = dst[ebase + row];                  // rows 0..15 (dup in hi half)

        f32x8 c[4];
        #pragma unroll
        for (int nt = 0; nt < 4; ++nt) c[nt] = (f32x8){0,0,0,0,0,0,0,0};
        #pragma unroll
        for (int nt = 0; nt < 4; ++nt)
            c[nt] = __builtin_amdgcn_wmma_f32_16x16x32_bf16(
                        false, a0.v, false, b[nt].v, (short)0, c[nt], false, false);
        #pragma unroll
        for (int nt = 0; nt < 4; ++nt)
            c[nt] = __builtin_amdgcn_wmma_f32_16x16x32_bf16(
                        false, a1.v, false, b[4 + nt].v, (short)0, c[nt], false, false);

        #pragma unroll
        for (int j = 0; j < 8; ++j) {
            int M = hi * 8 + j;
            int d = __shfl(myDst, M);                  // ds_bpermute: row M's dst
            int base = d * 64 + row;
            #pragma unroll
            for (int nt = 0; nt < 4; ++nt) {
                float val = fmaxf(c[nt][j] + bv[nt], 0.f);
                atomicAdd(&racc[base + nt * 16], val);
            }
        }
    }
}

// --------------------------- layer-1 node update ---------------------------
// Logical input x[n] = concat(h(bf16,64) | r1(f32,64) | [r2(f32,64)]), KT k-tiles.
static __device__ __forceinline__ uint4 load8_concat(
    const unsigned short* h, const float* r1, const float* r2, int n, int k0)
{
    if (k0 < 64) return *(const uint4*)(h + (long long)n * 64 + k0);
    const float* rp = (k0 < 128) ? (r1 + (long long)n * 64 + (k0 - 64))
                                 : (r2 + (long long)n * 64 + (k0 - 128));
    float4 f0 = *(const float4*)(rp);
    float4 f1 = *(const float4*)(rp + 4);
    uint4 u;
    u.x = (unsigned int)f2bf(f0.x) | ((unsigned int)f2bf(f0.y) << 16);
    u.y = (unsigned int)f2bf(f0.z) | ((unsigned int)f2bf(f0.w) << 16);
    u.z = (unsigned int)f2bf(f1.x) | ((unsigned int)f2bf(f1.y) << 16);
    u.w = (unsigned int)f2bf(f1.z) | ((unsigned int)f2bf(f1.w) << 16);
    return u;
}

__global__ __launch_bounds__(256) void node_update_kernel(
    const unsigned short* __restrict__ hseg,   // bf16 [N*64]
    const float* __restrict__ r1, const float* __restrict__ r2,
    const unsigned short* __restrict__ Wpk,    // bf16 [KT][4][32][16]
    const float* __restrict__ bias,            // [64]
    float* __restrict__ out,                   // f32 [N*64]
    int N, int KT)
{
    __shared__ unsigned short wlds[12288];     // up to 6 kt * 2048
    for (int i = threadIdx.x; i < KT * 2048; i += 256)
        wlds[i] = Wpk[i];
    __syncthreads();

    int lane = threadIdx.x & 31;
    int row  = lane & 15;
    int hi   = lane >> 4;
    float bv[4];
    #pragma unroll
    for (int nt = 0; nt < 4; ++nt) bv[nt] = bias[nt * 16 + row];

    int wave   = blockIdx.x * (blockDim.x >> 5) + (threadIdx.x >> 5);
    int nwaves = gridDim.x * (blockDim.x >> 5);
    int tiles  = N / 16;                       // N is a multiple of 16

    for (int tile = wave; tile < tiles; tile += nwaves) {   // wave-uniform loop
        int nbase = tile * 16;
        int n = nbase + row;

        f32x8 c[4];
        #pragma unroll
        for (int nt = 0; nt < 4; ++nt) c[nt] = (f32x8){0,0,0,0,0,0,0,0};

        for (int kt = 0; kt < KT; ++kt) {
            AFrag a;
            int k0 = kt * 32 + hi * 8;
            a.u[0] = load8_concat(hseg, r1, r2, n, k0);
            a.u[1] = load8_concat(hseg, r1, r2, n, k0 + 16);
            #pragma unroll
            for (int nt = 0; nt < 4; ++nt) {
                AFrag b;
                const unsigned short* wp = &wlds[((kt * 4 + nt) * 32 + lane) << 4];
                b.u[0] = *(const uint4*)(wp);
                b.u[1] = *(const uint4*)(wp + 8);
                c[nt] = __builtin_amdgcn_wmma_f32_16x16x32_bf16(
                            false, a.v, false, b.v, (short)0, c[nt], false, false);
            }
        }

        #pragma unroll
        for (int nt = 0; nt < 4; ++nt) {
            #pragma unroll
            for (int j = 0; j < 8; ++j) {
                int M = hi * 8 + j;
                out[(nbase + M) * 64 + nt * 16 + row] = fmaxf(c[nt][j] + bv[nt], 0.f);
            }
        }
    }
}

// --------------------------------- host ------------------------------------

static inline size_t al256(size_t x) { return (x + 255) & ~(size_t)255; }
static inline int imin(int a, int b) { return a < b ? a : b; }

extern "C" void kernel_launch(void* const* d_in, const int* in_sizes, int n_in,
                              void* d_out, int out_size, void* d_ws, size_t ws_size,
                              hipStream_t stream) {
    (void)in_sizes; (void)n_in; (void)out_size; (void)ws_size;

    const int*   src_aa = (const int*)d_in[0];
    const int*   dst_aa = (const int*)d_in[1];
    const int*   src_ab = (const int*)d_in[2];
    const int*   dst_ab = (const int*)d_in[3];
    const int*   src_ba = (const int*)d_in[4];
    const int*   dst_ba = (const int*)d_in[5];
    const float* emb_A  = (const float*)d_in[6];
    const float* emb_B  = (const float*)d_in[7];
    const float* W_aa   = (const float*)d_in[8];
    const float* b_aa   = (const float*)d_in[9];
    const float* W_ab   = (const float*)d_in[10];
    const float* b_ab   = (const float*)d_in[11];
    const float* W_ba   = (const float*)d_in[12];
    const float* b_ba   = (const float*)d_in[13];
    const float* Wn_A   = (const float*)d_in[14];
    const float* bn_A   = (const float*)d_in[15];
    const float* Wn_B   = (const float*)d_in[16];
    const float* bn_B   = (const float*)d_in[17];
    float* out = (float*)d_out;

    char* ws = (char*)d_ws;
    size_t off = 0;
    size_t o_deg_aa = off; off += al256((size_t)NA * 4);
    size_t o_deg_ab = off; off += al256((size_t)NB * 4);
    size_t o_deg_ba = off; off += al256((size_t)NA * 4);
    size_t o_r_aa   = off; off += al256((size_t)NA * 64 * 4);
    size_t o_r_ba   = off; off += al256((size_t)NA * 64 * 4);
    size_t o_r_ab   = off; off += al256((size_t)NB * 64 * 4);
    size_t zero_bytes = off;                       // degrees + accumulators
    size_t o_vecs   = off; off += al256(512 * 4);
    size_t o_hA1    = off; off += al256((size_t)NA * 64 * 2);
    size_t o_hB1    = off; off += al256((size_t)NB * 64 * 2);
    size_t o_Wpk_aa = off; off += al256(4096 * 2);
    size_t o_Wpk_ab = off; off += al256(4096 * 2);
    size_t o_Wpk_ba = off; off += al256(4096 * 2);
    size_t o_WpkA   = off; off += al256(12288 * 2);
    size_t o_WpkB   = off; off += al256(8192 * 2);

    int*   deg_aa = (int*)(ws + o_deg_aa);
    int*   deg_ab = (int*)(ws + o_deg_ab);
    int*   deg_ba = (int*)(ws + o_deg_ba);
    float* r_aa   = (float*)(ws + o_r_aa);
    float* r_ba   = (float*)(ws + o_r_ba);
    float* r_ab   = (float*)(ws + o_r_ab);
    float* vecs   = (float*)(ws + o_vecs);
    unsigned short* hA1    = (unsigned short*)(ws + o_hA1);
    unsigned short* hB1    = (unsigned short*)(ws + o_hB1);
    unsigned short* Wpk_aa = (unsigned short*)(ws + o_Wpk_aa);
    unsigned short* Wpk_ab = (unsigned short*)(ws + o_Wpk_ab);
    unsigned short* Wpk_ba = (unsigned short*)(ws + o_Wpk_ba);
    unsigned short* WpkA   = (unsigned short*)(ws + o_WpkA);
    unsigned short* WpkB   = (unsigned short*)(ws + o_WpkB);

    // 1) zero degrees + accumulators
    long long nzero = (long long)(zero_bytes / 4);
    zero_u32_kernel<<<4096, 256, 0, stream>>>((unsigned int*)ws, nzero);

    // 2) degree counts (layer 0 is analytic in degrees)
    degree_kernel<<<(EAA + 255) / 256, 256, 0, stream>>>(dst_aa, deg_aa, EAA);
    degree_kernel<<<(EAB + 255) / 256, 256, 0, stream>>>(dst_ab, deg_ab, EAB);
    degree_kernel<<<(EBA + 255) / 256, 256, 0, stream>>>(dst_ba, deg_ba, EBA);

    // 3) layer-0 precompute (tiny)
    layer0_precompute_kernel<<<1, 64, 0, stream>>>(
        emb_A, emb_B,
        W_aa, b_aa, W_ab, b_ab, W_ba, b_ba,
        Wn_A, bn_A, Wn_B, bn_B, vecs);

    // 4) layer-0 node features -> bf16
    layer0_nodeA_kernel<<<(NA * 64 + 255) / 256, 256, 0, stream>>>(vecs, deg_aa, deg_ba, hA1, NA);
    layer0_nodeB_kernel<<<(NB * 64 + 255) / 256, 256, 0, stream>>>(vecs, deg_ab, hB1, NB);

    // 5) pack layer-1 weights into WMMA B-fragment layout (bf16)
    pack_w_kernel<<<(2 * 2048 + 255) / 256, 256, 0, stream>>>(W_aa + 64 * 64, Wpk_aa, 2);
    pack_w_kernel<<<(2 * 2048 + 255) / 256, 256, 0, stream>>>(W_ab + 64 * 64, Wpk_ab, 2);
    pack_w_kernel<<<(2 * 2048 + 255) / 256, 256, 0, stream>>>(W_ba + 64 * 64, Wpk_ba, 2);
    pack_w_kernel<<<(6 * 2048 + 255) / 256, 256, 0, stream>>>(Wn_A + 192 * 64, WpkA, 6);
    pack_w_kernel<<<(4 * 2048 + 255) / 256, 256, 0, stream>>>(Wn_B + 128 * 64, WpkB, 4);

    // 6) layer-1 edge messages (persistent WMMA waves) + f32 atomic scatter
    {
        int tiles = EAA / 16, blocks = imin((tiles + 7) / 8, 1280);
        edge_msg_kernel<<<blocks, 256, 0, stream>>>(hA1, src_aa, dst_aa, Wpk_aa, b_aa + 64, r_aa, tiles);
    }
    {
        int tiles = EAB / 16, blocks = imin((tiles + 7) / 8, 1280);
        edge_msg_kernel<<<blocks, 256, 0, stream>>>(hA1, src_ab, dst_ab, Wpk_ab, b_ab + 64, r_ab, tiles);
    }
    {
        int tiles = EBA / 16, blocks = imin((tiles + 7) / 8, 1280);
        edge_msg_kernel<<<blocks, 256, 0, stream>>>(hB1, src_ba, dst_ba, Wpk_ba, b_ba + 64, r_ba, tiles);
    }

    // 7) layer-1 node updates (WMMA over concat) -> final output
    {
        int tiles = NA / 16, blocks = imin((tiles + 7) / 8, 1280);
        node_update_kernel<<<blocks, 256, 0, stream>>>(hA1, r_aa, r_ba, WpkA, bn_A + 64, out, NA, 6);
    }
    {
        int tiles = NB / 16, blocks = imin((tiles + 7) / 8, 1280);
        node_update_kernel<<<blocks, 256, 0, stream>>>(hB1, r_ab, (const float*)nullptr, WpkB, bn_B + 64,
                                                       out + (long long)NA * 64, NB, 4);
    }
}
